// EnBaseLayer_48576080117843
// MI455X (gfx1250) — matrix operand order repaired and checked by reading, not compiled, not run on hardware
//
#include <hip/hip_runtime.h>
#include <hip/hip_bf16.h>

// ---------------------------------------------------------------------------
// EnBaseLayer fused WMMA implementation for gfx1250 (MI455X).
//   N=40000 nodes, E=640000 edges, H=128, EF=8, IN_E=284 (pad to 288).
//   bf16 WMMA (v_wmma_f32_16x16x32_bf16), f32 accumulate.
//   Each wave handles TWO 16-row M-tiles per iteration so every B fragment
//   loaded from LDS feeds two WMMAs (halves LDS traffic per matrix op).
// ---------------------------------------------------------------------------

typedef __attribute__((ext_vector_type(16))) __bf16 v16bf;
typedef __attribute__((ext_vector_type(8)))  __bf16 v8bf;
typedef __attribute__((ext_vector_type(8)))  float  v8f;
typedef __attribute__((ext_vector_type(4)))  float  v4f;

#define N_NODES 40000
#define N_EDGES 640000
#define HD 128
#define EF 8
#define IN_E 284
#define KP1 288           // padded K for layer-1 (9 tiles of 32)
#define NT 8              // 128 / 16 output tiles
#define MT 2              // M-tiles (of 16 rows) per wave iteration
#define WPB 8             // waves per block
#define THREADS 256

__constant__ float c_offset[20] = {0.f,1.f,1.25f,1.5f,1.75f,2.f,2.25f,2.5f,2.75f,3.f,
                                   3.5f,4.f,4.5f,5.f,5.5f,6.f,7.f,8.f,9.f,10.f};

struct WaveScr {            // per-wave LDS scratch for one 32-edge tile
    int   dstn[32];
    int   srcn[32];
    float relx[96];
    float dist[32];
};                          // 768 B

__device__ __forceinline__ float siluf(float x)  { return x / (1.f + __expf(-x)); }
__device__ __forceinline__ float sigmf(float x)  { return 1.f / (1.f + __expf(-x)); }

__device__ __forceinline__ v16bf cat8(v8bf lo, v8bf hi) {
    return __builtin_shufflevector(lo, hi, 0,1,2,3,4,5,6,7,8,9,10,11,12,13,14,15);
}
__device__ __forceinline__ v8f wmma_bf16(v16bf a, v16bf b, v8f c) {
    return __builtin_amdgcn_wmma_f32_16x16x32_bf16(false, a, false, b, (short)0, c,
                                                   false, false);
}
// 16x32 bf16 A-fragment from a row-major f32 row (32 cols starting at rb).
// half h2==0 lanes hold K {0..7,16..23}; h2==1 lanes hold K {8..15,24..31}.
__device__ __forceinline__ v16bf a_frag_f32(const float* rb, int h2) {
    const float* p0 = rb + h2 * 8;
    const float* p1 = rb + 16 + h2 * 8;
    v4f f0 = *(const v4f*)p0, f1 = *(const v4f*)(p0 + 4);
    v4f f2 = *(const v4f*)p1, f3 = *(const v4f*)(p1 + 4);
    v16bf a;
#pragma unroll
    for (int i = 0; i < 4; i++) {
        a[i]      = (__bf16)f0[i];
        a[4 + i]  = (__bf16)f1[i];
        a[8 + i]  = (__bf16)f2[i];
        a[12 + i] = (__bf16)f3[i];
    }
    return a;
}
// B-fragment (32x16 bf16): lane holds N column lm; half h2 holds
// K {kt*32 + h2*8 + 0..7, kt*32 + 16 + h2*8 + 0..7} (strided by HD).
__device__ __forceinline__ v16bf b_frag(const __bf16* wrow_ncol) {
    v16bf b;
#pragma unroll
    for (int i = 0; i < 8; i++) {
        b[i]     = wrow_ncol[i * HD];
        b[8 + i] = wrow_ncol[(16 + i) * HD];
    }
    return b;
}

// ---------------------------------------------------------------------------
__global__ void init_kernel(const float* __restrict__ x, float* __restrict__ mi,
                            float* __restrict__ x_out) {
    size_t i = (size_t)blockIdx.x * blockDim.x + threadIdx.x;
    if (i < (size_t)N_NODES * HD) mi[i] = 0.f;
    if (i < (size_t)N_NODES * 3)  x_out[i] = x[i];
}

// ---------------------------------------------------------------------------
__global__ __launch_bounds__(THREADS)
void edge_kernel(const float* __restrict__ hbuf, const float* __restrict__ x,
                 const int* __restrict__ eidx, const float* __restrict__ eattr,
                 const float* __restrict__ We1, const float* __restrict__ be1,
                 const float* __restrict__ We2, const float* __restrict__ be2,
                 const float* __restrict__ Winf, const float* __restrict__ binf,
                 const float* __restrict__ Wx1, const float* __restrict__ bx1,
                 const float* __restrict__ Wx2,
                 float* __restrict__ mi, float* __restrict__ x_out) {
    extern __shared__ char smem[];
    __bf16* sWe1 = (__bf16*)smem;                 // 288*128 bf16
    __bf16* sWe2 = sWe1 + KP1 * HD;               // 128*128
    __bf16* sWx1 = sWe2 + HD * HD;                // 128*128
    float*  sBe1 = (float*)(sWx1 + HD * HD);      // 128 f32 each
    float*  sBe2 = sBe1 + HD;
    float*  sBx1 = sBe2 + HD;
    float*  sWin = sBx1 + HD;
    float*  sWx2 = sWin + HD;
    WaveScr* wscr = (WaveScr*)(sWx2 + HD);        // WPB structs
    __bf16* smbuf = (__bf16*)(wscr + WPB);        // WPB * 32*128 bf16

    const int tid = threadIdx.x;
    for (int i = tid; i < KP1 * HD; i += THREADS) {
        int k = i >> 7, n = i & (HD - 1);
        sWe1[i] = (__bf16)((k < IN_E) ? We1[k * HD + n] : 0.f);
    }
    for (int i = tid; i < HD * HD; i += THREADS) sWe2[i] = (__bf16)We2[i];
    for (int i = tid; i < HD * HD; i += THREADS) sWx1[i] = (__bf16)Wx1[i];
    for (int i = tid; i < HD; i += THREADS) {
        sBe1[i] = be1[i]; sBe2[i] = be2[i]; sBx1[i] = bx1[i];
        sWin[i] = Winf[i]; sWx2[i] = Wx2[i];
    }
    __syncthreads();
    const float binf0 = binf[0];

    const int wave = tid >> 5, lane = tid & 31;
    const int h2 = lane >> 4, lm = lane & 15;
    WaveScr* wsc  = wscr + wave;
    __bf16*  mbuf = smbuf + wave * (32 * HD);

    const int nPairs  = N_EDGES / (16 * MT);      // 20000
    const int gw      = blockIdx.x * WPB + wave;
    const int wstride = gridDim.x * WPB;

    for (int pair = gw; pair < nPairs; pair += wstride) {
        const int e0 = pair * 32;
        {                                          // per-edge setup, all 32 lanes
            int e = e0 + lane;
            int s = eidx[e], d = eidx[N_EDGES + e];
            wsc->srcn[lane] = s; wsc->dstn[lane] = d;
            float rx = x[d*3+0]-x[s*3+0], ry = x[d*3+1]-x[s*3+1], rz = x[d*3+2]-x[s*3+2];
            float dist = __builtin_sqrtf(rx*rx + ry*ry + rz*rz + 1e-8f);
            wsc->relx[lane*3+0]=rx; wsc->relx[lane*3+1]=ry; wsc->relx[lane*3+2]=rz;
            wsc->dist[lane] = dist;
        }
        asm volatile("s_wait_dscnt 0x0" ::: "memory");   // intra-wave DS RAW

        int rdst[MT], rsrc[MT]; float rdist[MT];
#pragma unroll
        for (int mt = 0; mt < MT; mt++) {
            rdst[mt]  = wsc->dstn[mt*16 + lm];
            rsrc[mt]  = wsc->srcn[mt*16 + lm];
            rdist[mt] = wsc->dist[mt*16 + lm];
        }

        // ---- layer 1: [32x288] @ We1[288x128] ------------------------------
        v8f acc[MT][NT];
#pragma unroll
        for (int mt = 0; mt < MT; mt++)
#pragma unroll
            for (int nt = 0; nt < NT; nt++) { v8f z = {}; acc[mt][nt] = z; }

#pragma unroll
        for (int kt = 0; kt < 9; kt++) {
            v16bf a[MT];
            if (kt < 4) {
#pragma unroll
                for (int mt = 0; mt < MT; mt++)
                    a[mt] = a_frag_f32(hbuf + (size_t)rdst[mt] * HD + kt * 32, h2);
            } else if (kt < 8) {
#pragma unroll
                for (int mt = 0; mt < MT; mt++)
                    a[mt] = a_frag_f32(hbuf + (size_t)rsrc[mt] * HD + (kt - 4) * 32, h2);
            } else {                               // K 256..287: gauss|attr|pad
#pragma unroll
                for (int mt = 0; mt < MT; mt++) {
#pragma unroll
                    for (int i = 0; i < 16; i++) {
                        int c = (i < 8) ? (h2*8 + i) : (16 + h2*8 + (i - 8));
                        float v;
                        if (c < 20) { float dd = rdist[mt] - c_offset[c];
                                      v = __expf(-0.5f * dd * dd); }
                        else if (c < 28)
                            v = eattr[(size_t)(e0 + mt*16 + lm) * EF + (c - 20)];
                        else v = 0.f;
                        a[mt][i] = (__bf16)v;
                    }
                }
            }
            const __bf16* wk = sWe1 + (size_t)(kt*32 + h2*8) * HD + lm;
#pragma unroll
            for (int nt = 0; nt < NT; nt++) {
                v16bf b = b_frag(wk + nt * 16);    // one B feeds both M-tiles
                acc[0][nt] = wmma_bf16(a[0], b, acc[0][nt]);
                acc[1][nt] = wmma_bf16(a[1], b, acc[1][nt]);
            }
        }
        // bias + silu -> mbuf (bf16, D-layout -> row-major, 32 rows)
#pragma unroll
        for (int nt = 0; nt < NT; nt++) {
            float b = sBe1[nt*16 + lm];
#pragma unroll
            for (int mt = 0; mt < MT; mt++)
#pragma unroll
                for (int r = 0; r < 8; r++)
                    mbuf[(mt*16 + h2*8 + r) * HD + nt*16 + lm] =
                        (__bf16)siluf(acc[mt][nt][r] + b);
        }
        asm volatile("s_wait_dscnt 0x0" ::: "memory");

        // ---- layer 2: mij @ We2 -------------------------------------------
        v16bf aM[MT][4];
#pragma unroll
        for (int mt = 0; mt < MT; mt++)
#pragma unroll
            for (int kt = 0; kt < 4; kt++) {
                const __bf16* rb = mbuf + (mt*16 + lm) * HD + kt * 32 + h2 * 8;
                aM[mt][kt] = cat8(*(const v8bf*)rb, *(const v8bf*)(rb + 16));
            }
        v8f m2[MT][NT];
#pragma unroll
        for (int nt = 0; nt < NT; nt++) {
            v8f c0 = {}, c1 = {};
#pragma unroll
            for (int kt = 0; kt < 4; kt++) {
                v16bf b = b_frag(sWe2 + (size_t)(kt*32 + h2*8)*HD + nt*16 + lm);
                c0 = wmma_bf16(aM[0][kt], b, c0);
                c1 = wmma_bf16(aM[1][kt], b, c1);
            }
            float bb = sBe2[nt*16 + lm];
#pragma unroll
            for (int r = 0; r < 8; r++) { c0[r] = siluf(c0[r] + bb);
                                          c1[r] = siluf(c1[r] + bb); }
            m2[0][nt] = c0; m2[1][nt] = c1;
        }

        // ---- eij = sigmoid(m2 @ Winf + binf) (dot + 16-lane xor reduce) ----
        float esum[MT][8];
#pragma unroll
        for (int mt = 0; mt < MT; mt++)
#pragma unroll
            for (int r = 0; r < 8; r++) esum[mt][r] = 0.f;
#pragma unroll
        for (int nt = 0; nt < NT; nt++) {
            float wv = sWin[nt*16 + lm];
#pragma unroll
            for (int mt = 0; mt < MT; mt++)
#pragma unroll
                for (int r = 0; r < 8; r++) esum[mt][r] += m2[mt][nt][r] * wv;
        }
#pragma unroll
        for (int msk = 1; msk < 16; msk <<= 1)
#pragma unroll
            for (int mt = 0; mt < MT; mt++)
#pragma unroll
                for (int r = 0; r < 8; r++)
                    esum[mt][r] += __shfl_xor(esum[mt][r], msk, 32);

        int rowdst[MT][8];
#pragma unroll
        for (int mt = 0; mt < MT; mt++)
#pragma unroll
            for (int r = 0; r < 8; r++) rowdst[mt][r] = wsc->dstn[mt*16 + h2*8 + r];
        // ---- mi += mij * eij (f32 atomics into L2) -------------------------
#pragma unroll
        for (int mt = 0; mt < MT; mt++)
#pragma unroll
            for (int nt = 0; nt < NT; nt++)
#pragma unroll
                for (int r = 0; r < 8; r++) {
                    float e = sigmf(esum[mt][r] + binf0);
                    atomicAdd(mi + (size_t)rowdst[mt][r] * HD + nt*16 + lm,
                              m2[mt][nt][r] * e);
                }

        // ---- coordinate path: tanh(silu(m2@Wx1+bx1)@Wx2) ------------------
#pragma unroll
        for (int mt = 0; mt < MT; mt++)
#pragma unroll
            for (int nt = 0; nt < NT; nt++)
#pragma unroll
                for (int r = 0; r < 8; r++)
                    mbuf[(mt*16 + h2*8 + r) * HD + nt*16 + lm] = (__bf16)m2[mt][nt][r];
        asm volatile("s_wait_dscnt 0x0" ::: "memory");
#pragma unroll
        for (int mt = 0; mt < MT; mt++)
#pragma unroll
            for (int kt = 0; kt < 4; kt++) {
                const __bf16* rb = mbuf + (mt*16 + lm) * HD + kt * 32 + h2 * 8;
                aM[mt][kt] = cat8(*(const v8bf*)rb, *(const v8bf*)(rb + 16));
            }
        float xsum[MT][8];
#pragma unroll
        for (int mt = 0; mt < MT; mt++)
#pragma unroll
            for (int r = 0; r < 8; r++) xsum[mt][r] = 0.f;
#pragma unroll
        for (int nt = 0; nt < NT; nt++) {
            v8f c0 = {}, c1 = {};
#pragma unroll
            for (int kt = 0; kt < 4; kt++) {
                v16bf b = b_frag(sWx1 + (size_t)(kt*32 + h2*8)*HD + nt*16 + lm);
                c0 = wmma_bf16(aM[0][kt], b, c0);
                c1 = wmma_bf16(aM[1][kt], b, c1);
            }
            float bb = sBx1[nt*16 + lm], wv = sWx2[nt*16 + lm];
#pragma unroll
            for (int r = 0; r < 8; r++) { xsum[0][r] += siluf(c0[r] + bb) * wv;
                                          xsum[1][r] += siluf(c1[r] + bb) * wv; }
        }
#pragma unroll
        for (int msk = 1; msk < 16; msk <<= 1)
#pragma unroll
            for (int mt = 0; mt < MT; mt++)
#pragma unroll
                for (int r = 0; r < 8; r++)
                    xsum[mt][r] += __shfl_xor(xsum[mt][r], msk, 32);

        {   // one lane per edge row (32 rows, 32 lanes); cndmask select tree
            int mt_sel = lm >> 3, r_sel = lm & 7;
            int row = mt_sel*16 + h2*8 + r_sel;
            float xv = 0.f;
#pragma unroll
            for (int mt = 0; mt < MT; mt++)
#pragma unroll
                for (int r = 0; r < 8; r++)
                    xv = (mt == mt_sel && r == r_sel) ? xsum[mt][r] : xv;
            float xw = tanhf(xv);
            float s  = xw / (wsc->dist[row] + 1.f);
            int   dd = wsc->dstn[row];
            atomicAdd(x_out + (size_t)dd*3 + 0, wsc->relx[row*3+0] * s);
            atomicAdd(x_out + (size_t)dd*3 + 1, wsc->relx[row*3+1] * s);
            atomicAdd(x_out + (size_t)dd*3 + 2, wsc->relx[row*3+2] * s);
        }
    }
}

// ---------------------------------------------------------------------------
__global__ __launch_bounds__(THREADS)
void node_kernel(const float* __restrict__ hbuf, const float* __restrict__ mi,
                 const float* __restrict__ Wn1, const float* __restrict__ bn1,
                 const float* __restrict__ Wn2, const float* __restrict__ bn2,
                 float* __restrict__ h_out) {
    extern __shared__ char smem[];
    __bf16* sW1 = (__bf16*)smem;                  // 256*128
    __bf16* sW2 = sW1 + 256 * HD;                 // 128*128
    float*  sB1 = (float*)(sW2 + HD * HD);
    float*  sB2 = sB1 + HD;
    __bf16* smbuf = (__bf16*)(sB2 + HD);          // WPB * 32*128

    const int tid = threadIdx.x;
    for (int i = tid; i < 256 * HD; i += THREADS) sW1[i] = (__bf16)Wn1[i];
    for (int i = tid; i < HD * HD; i += THREADS)  sW2[i] = (__bf16)Wn2[i];
    for (int i = tid; i < HD; i += THREADS) { sB1[i] = bn1[i]; sB2[i] = bn2[i]; }
    __syncthreads();

    const int wave = tid >> 5, lane = tid & 31;
    const int h2 = lane >> 4, lm = lane & 15;
    __bf16* mbuf = smbuf + wave * (32 * HD);

    const int nPairs  = N_NODES / (16 * MT);      // 1250
    const int gw      = blockIdx.x * WPB + wave;
    const int wstride = gridDim.x * WPB;

    for (int pair = gw; pair < nPairs; pair += wstride) {
        const int n0 = pair * 32;
        v8f acc[MT][NT];
#pragma unroll
        for (int mt = 0; mt < MT; mt++)
#pragma unroll
            for (int nt = 0; nt < NT; nt++) { v8f z = {}; acc[mt][nt] = z; }
#pragma unroll
        for (int kt = 0; kt < 8; kt++) {
            v16bf a[MT];
#pragma unroll
            for (int mt = 0; mt < MT; mt++) {
                const float* rb = (kt < 4)
                    ? (mi   + (size_t)(n0 + mt*16 + lm) * HD + kt * 32)
                    : (hbuf + (size_t)(n0 + mt*16 + lm) * HD + (kt - 4) * 32);
                a[mt] = a_frag_f32(rb, h2);
            }
            const __bf16* wk = sW1 + (size_t)(kt*32 + h2*8) * HD + lm;
#pragma unroll
            for (int nt = 0; nt < NT; nt++) {
                v16bf b = b_frag(wk + nt * 16);
                acc[0][nt] = wmma_bf16(a[0], b, acc[0][nt]);
                acc[1][nt] = wmma_bf16(a[1], b, acc[1][nt]);
            }
        }
#pragma unroll
        for (int nt = 0; nt < NT; nt++) {
            float b = sB1[nt*16 + lm];
#pragma unroll
            for (int mt = 0; mt < MT; mt++)
#pragma unroll
                for (int r = 0; r < 8; r++)
                    mbuf[(mt*16 + h2*8 + r) * HD + nt*16 + lm] =
                        (__bf16)siluf(acc[mt][nt][r] + b);
        }
        asm volatile("s_wait_dscnt 0x0" ::: "memory");
        v16bf aM[MT][4];
#pragma unroll
        for (int mt = 0; mt < MT; mt++)
#pragma unroll
            for (int kt = 0; kt < 4; kt++) {
                const __bf16* rb = mbuf + (mt*16 + lm) * HD + kt * 32 + h2 * 8;
                aM[mt][kt] = cat8(*(const v8bf*)rb, *(const v8bf*)(rb + 16));
            }
#pragma unroll
        for (int nt = 0; nt < NT; nt++) {
            v8f c0 = {}, c1 = {};
#pragma unroll
            for (int kt = 0; kt < 4; kt++) {
                v16bf b = b_frag(sW2 + (size_t)(kt*32 + h2*8)*HD + nt*16 + lm);
                c0 = wmma_bf16(aM[0][kt], b, c0);
                c1 = wmma_bf16(aM[1][kt], b, c1);
            }
            float b = sB2[nt*16 + lm];
#pragma unroll
            for (int r = 0; r < 8; r++) {
                size_t o0 = (size_t)(n0 +      h2*8 + r) * HD + nt*16 + lm;
                size_t o1 = (size_t)(n0 + 16 + h2*8 + r) * HD + nt*16 + lm;
                h_out[o0] = hbuf[o0] + c0[r] + b;   // residual
                h_out[o1] = hbuf[o1] + c1[r] + b;
            }
        }
    }
}

// ---------------------------------------------------------------------------
extern "C" void kernel_launch(void* const* d_in, const int* in_sizes, int n_in,
                              void* d_out, int out_size, void* d_ws, size_t ws_size,
                              hipStream_t stream) {
    const float* hbuf = (const float*)d_in[0];
    const float* x    = (const float*)d_in[1];
    const int*   eidx = (const int*)d_in[2];
    // d_in[3] mask_ligand unused by the reference
    const float* eattr = (const float*)d_in[4];
    const float* We1 = (const float*)d_in[5],  *be1 = (const float*)d_in[6];
    const float* We2 = (const float*)d_in[7],  *be2 = (const float*)d_in[8];
    const float* Winf= (const float*)d_in[9],  *binf= (const float*)d_in[10];
    const float* Wx1 = (const float*)d_in[11], *bx1 = (const float*)d_in[12];
    const float* Wx2 = (const float*)d_in[13];
    const float* Wn1 = (const float*)d_in[14], *bn1 = (const float*)d_in[15];
    const float* Wn2 = (const float*)d_in[16], *bn2 = (const float*)d_in[17];

    float* h_out = (float*)d_out;
    float* x_out = (float*)d_out + (size_t)N_NODES * HD;
    float* mi    = (float*)d_ws;                 // N*128 f32 accumulator

    const size_t lds_edge = (size_t)KP1*HD*2 + 2*(size_t)HD*HD*2 + 5*HD*4
                          + WPB*sizeof(WaveScr) + (size_t)WPB*32*HD*2;  // ~208 KB
    const size_t lds_node = (size_t)256*HD*2 + (size_t)HD*HD*2 + 2*HD*4
                          + (size_t)WPB*32*HD*2;                         // ~161 KB
    (void)hipFuncSetAttribute((const void*)edge_kernel,
        hipFuncAttributeMaxDynamicSharedMemorySize, (int)lds_edge);
    (void)hipFuncSetAttribute((const void*)node_kernel,
        hipFuncAttributeMaxDynamicSharedMemorySize, (int)lds_node);

    init_kernel<<<(N_NODES * HD + THREADS - 1) / THREADS, THREADS, 0, stream>>>(
        x, mi, x_out);
    edge_kernel<<<512, THREADS, lds_edge, stream>>>(
        hbuf, x, eidx, eattr, We1, be1, We2, be2, Winf, binf, Wx1, bx1, Wx2,
        mi, x_out);
    node_kernel<<<160, THREADS, lds_node, stream>>>(
        hbuf, mi, Wn1, bn1, Wn2, bn2, h_out);
}
